// Inflector_35158602285790
// MI455X (gfx1250) — matrix operand order repaired and checked by reading, not compiled, not run on hardware
//
#include <hip/hip_runtime.h>
#include <cstdint>

// ---------------------------------------------------------------------------
// Types for CDNA5 WMMA (wave32): v_wmma_f32_16x16x32_bf16
// ---------------------------------------------------------------------------
typedef __bf16 bf16_t;
typedef __attribute__((ext_vector_type(8)))  bf16_t v8bf;
typedef __attribute__((ext_vector_type(16))) bf16_t v16bf;
typedef __attribute__((ext_vector_type(8)))  float  v8f;

// model dimensions (fixed by the reference)
constexpr int E   = 256;
constexpr int H   = 512;
constexpr int V   = 1024;
constexpr int OPS = 4;
constexpr int Bz  = 512;
constexpr int Lz  = 48;
constexpr int Sz  = 24;
constexpr int Tz  = 40;             // TDEC
constexpr int H4  = 4 * H;          // 2048
constexpr int KDEC = E + 2 * H + H; // 1792 : [x | seq_attn | tag_attn | h]
constexpr float EPSF = 1e-14f;

static __device__ __forceinline__ float sigf(float x) {
  return 1.0f / (1.0f + __expf(-x));
}

static __device__ __forceinline__ float wave_reduce_sum(float v) {
  #pragma unroll
  for (int m = 16; m >= 1; m >>= 1) v += __shfl_xor(v, m, 32);
  return v;
}

static __device__ __forceinline__ v16bf frag16(const bf16_t* p0, const bf16_t* p1) {
  v8bf lo = *(const v8bf*)p0;
  v8bf hi = *(const v8bf*)p1;
  return __builtin_shufflevector(lo, hi, 0,1,2,3,4,5,6,7,8,9,10,11,12,13,14,15);
}

// ---------------------------------------------------------------------------
// WMMA GEMM:  C[M,N] = act( A[M,K](f32) * W[N,K]^T(bf16) + bias + addend )
// Block tile 64x128, 8 waves (2x4), each wave owns a 32x32 C region =
// 4 independent 16x16 f32 accumulators -> 4 v_wmma per K-chunk (A/B reuse).
//
// Manually software-pipelined with double-buffered LDS tiles:
//   - W tile: GLOBAL_LOAD_ASYNC_TO_LDS_B128 issued one chunk ahead; the
//     s_wait_asynccnt 2 before each barrier lets the 2 newest (next-buffer)
//     loads stay in flight while guaranteeing the current buffer landed
//     (async loads complete in order per ISA 5.7.1).
//   - A tile: global f32 loads issued one chunk ahead into VGPRs, converted
//     to bf16 and stored to the other LDS buffer after the current compute.
//   - One workgroup barrier per K-chunk.
// Requires: M % 64 == 0, N % 128 == 0, K % 32 == 0 (true for every GEMM here).
// ACT: 0=none, 1=relu, 2=leaky(0.01)
// ---------------------------------------------------------------------------
template <int ACT>
__global__ __launch_bounds__(256)
void k_gemm(const float* __restrict__ A, int lda,
            const bf16_t* __restrict__ W,
            const float* __restrict__ bias,
            const float* __restrict__ addend, int ldadd,
            float* __restrict__ C, int ldc,
            int M, int N, int K)
{
  constexpr int BM = 64, BN = 128, BK = 32, LDT = BK + 8;
  constexpr int TILE = (BM + BN) * LDT;   // elements per buffer
  // single static LDS allocation -> starts at LDS offset 0, so byte offsets
  // for the async-load VDST operands are well-defined.
  __shared__ bf16_t smem[2 * TILE];

  const int m0   = blockIdx.y * BM;
  const int n0   = blockIdx.x * BN;
  const int tid  = threadIdx.x;
  const int lane = tid & 31;
  const int wave = tid >> 5;
  const int wm   = wave >> 2;   // 0..1 : rows wm*32 .. +31
  const int wn   = wave & 3;    // 0..3 : cols wn*32 .. +31
  const int hi   = lane >> 4;   // lane group 0/1
  const int l16  = lane & 15;

  v8f acc00 = {0.f,0.f,0.f,0.f,0.f,0.f,0.f,0.f};
  v8f acc01 = {0.f,0.f,0.f,0.f,0.f,0.f,0.f,0.f};
  v8f acc10 = {0.f,0.f,0.f,0.f,0.f,0.f,0.f,0.f};
  v8f acc11 = {0.f,0.f,0.f,0.f,0.f,0.f,0.f,0.f};

  // staging coordinates (per thread, constant over the K loop)
  const int ar_ = tid >> 2, acs = (tid & 3) * 8;   // A: 64 rows x 32 f32, 8/thread
  const int wr_ = tid >> 1, wcs = (tid & 1) * 16;  // W: 128 rows x 32 bf16, 16/thread
  const float*  Aptr = A + (size_t)(m0 + ar_) * lda + acs;
  const bf16_t* Wptr = W + (size_t)(n0 + wr_) * K + wcs;
  const unsigned wlds0     = (unsigned)((BM + wr_) * LDT + wcs) * 2u; // bytes, buffer 0
  const unsigned tilebytes = (unsigned)TILE * 2u;

  // ---- prologue: stage chunk 0 into buffer 0 ----
  asm volatile("global_load_async_to_lds_b128 %0, %1, off"
               :: "v"(wlds0), "v"(Wptr) : "memory");
  asm volatile("global_load_async_to_lds_b128 %0, %1, off"
               :: "v"(wlds0 + 16u), "v"(Wptr + 8) : "memory");
  {
    float4 v0 = *(const float4*)(Aptr);
    float4 v1 = *(const float4*)(Aptr + 4);
    v8bf p;
    p[0] = (bf16_t)v0.x; p[1] = (bf16_t)v0.y;
    p[2] = (bf16_t)v0.z; p[3] = (bf16_t)v0.w;
    p[4] = (bf16_t)v1.x; p[5] = (bf16_t)v1.y;
    p[6] = (bf16_t)v1.z; p[7] = (bf16_t)v1.w;
    *(v8bf*)&smem[ar_ * LDT + acs] = p;
  }

  int p = 0;
  for (int kc = 0; kc < K; kc += BK) {
    const bool more = (kc + BK) < K;
    float4 nv0 = make_float4(0.f, 0.f, 0.f, 0.f);
    float4 nv1 = make_float4(0.f, 0.f, 0.f, 0.f);
    if (more) {
      // issue next chunk: async W -> other buffer, A f32 -> VGPRs
      const unsigned wl = wlds0 + (unsigned)(1 - p) * tilebytes;
      asm volatile("global_load_async_to_lds_b128 %0, %1, off"
                   :: "v"(wl), "v"(Wptr + kc + BK) : "memory");
      asm volatile("global_load_async_to_lds_b128 %0, %1, off"
                   :: "v"(wl + 16u), "v"(Wptr + kc + BK + 8) : "memory");
      nv0 = *(const float4*)(Aptr + kc + BK);
      nv1 = *(const float4*)(Aptr + kc + BK + 4);
      // current buffer's W loads (issued last iteration, in-order) are done
      // once only the 2 just-issued remain outstanding.
      asm volatile("s_wait_asynccnt 2" ::: "memory");
    } else {
      asm volatile("s_wait_asynccnt 0" ::: "memory");
    }
    __syncthreads();

    const bf16_t* Ab = smem + p * TILE;
    const bf16_t* Wb = Ab + BM * LDT;

    // A fragments per ISA 16-bit 16x32 layout:
    //  lanes 0-15 : halves 0..7 = K 0..7,  halves 8..15 = K 16..23
    //  lanes 16-31: halves 0..7 = K 8..15, halves 8..15 = K 24..31
    const int ar0 = wm * 32 + l16;
    const int ar1 = ar0 + 16;
    v16bf a0 = frag16(&Ab[ar0 * LDT + hi * 8], &Ab[ar0 * LDT + 16 + hi * 8]);
    v16bf a1 = frag16(&Ab[ar1 * LDT + hi * 8], &Ab[ar1 * LDT + 16 + hi * 8]);

    // B fragments (32x16): same K layout as A (lanes 0-15: K 0..7,16..23; lanes 16-31: K 8..15,24..31)
    const int bc0 = wn * 32 + l16;
    const int bc1 = bc0 + 16;
    v16bf b0 = frag16(&Wb[bc0 * LDT + hi * 8], &Wb[bc0 * LDT + 16 + hi * 8]);
    v16bf b1 = frag16(&Wb[bc1 * LDT + hi * 8], &Wb[bc1 * LDT + 16 + hi * 8]);

    acc00 = __builtin_amdgcn_wmma_f32_16x16x32_bf16(false, a0, false, b0, (short)0, acc00, false, false);
    acc01 = __builtin_amdgcn_wmma_f32_16x16x32_bf16(false, a0, false, b1, (short)0, acc01, false, false);
    acc10 = __builtin_amdgcn_wmma_f32_16x16x32_bf16(false, a1, false, b0, (short)0, acc10, false, false);
    acc11 = __builtin_amdgcn_wmma_f32_16x16x32_bf16(false, a1, false, b1, (short)0, acc11, false, false);

    if (more) {
      // convert and park next A chunk in the other buffer (read next iter)
      v8bf pk;
      pk[0] = (bf16_t)nv0.x; pk[1] = (bf16_t)nv0.y;
      pk[2] = (bf16_t)nv0.z; pk[3] = (bf16_t)nv0.w;
      pk[4] = (bf16_t)nv1.x; pk[5] = (bf16_t)nv1.y;
      pk[6] = (bf16_t)nv1.z; pk[7] = (bf16_t)nv1.w;
      *(v8bf*)&smem[(1 - p) * TILE + ar_ * LDT + acs] = pk;
    }
    p ^= 1;
  }

  // C layout: lanes 0-15 -> M = vgpr, lanes 16-31 -> M = vgpr + 8 ; N = lane&15
  const int colA = n0 + wn * 32 + l16;
  const int colB = colA + 16;
  const float bvA = bias ? bias[colA] : 0.f;
  const float bvB = bias ? bias[colB] : 0.f;
  const int rb0 = m0 + wm * 32 + hi * 8;
  const int rb1 = rb0 + 16;
  #pragma unroll
  for (int v = 0; v < 8; ++v) {
    const int row0 = rb0 + v;
    const int row1 = rb1 + v;
    float x00 = acc00[v] + bvA, x01 = acc01[v] + bvB;
    float x10 = acc10[v] + bvA, x11 = acc11[v] + bvB;
    if (addend) {
      x00 += addend[(size_t)row0 * ldadd + colA];
      x01 += addend[(size_t)row0 * ldadd + colB];
      x10 += addend[(size_t)row1 * ldadd + colA];
      x11 += addend[(size_t)row1 * ldadd + colB];
    }
    if (ACT == 1) {
      x00 = fmaxf(x00, 0.f); x01 = fmaxf(x01, 0.f);
      x10 = fmaxf(x10, 0.f); x11 = fmaxf(x11, 0.f);
    }
    if (ACT == 2) {
      x00 = (x00 > 0.f) ? x00 : 0.01f * x00;
      x01 = (x01 > 0.f) ? x01 : 0.01f * x01;
      x10 = (x10 > 0.f) ? x10 : 0.01f * x10;
      x11 = (x11 > 0.f) ? x11 : 0.01f * x11;
    }
    C[(size_t)row0 * ldc + colA] = x00;
    C[(size_t)row0 * ldc + colB] = x01;
    C[(size_t)row1 * ldc + colA] = x10;
    C[(size_t)row1 * ldc + colB] = x11;
  }
}

// ---------------------------------------------------------------------------
// small utility kernels
// ---------------------------------------------------------------------------
__global__ void k_cvt_bf16(const float* __restrict__ s, bf16_t* __restrict__ d, int n) {
  int i = blockIdx.x * 256 + threadIdx.x;
  if (i < n) d[i] = (bf16_t)s[i];
}

__global__ void k_add2(const float* __restrict__ a, const float* __restrict__ b,
                       float* __restrict__ o, int n) {
  int i = blockIdx.x * 256 + threadIdx.x;
  if (i < n) o[i] = a[i] + b[i];
}

// decoder fused weight: dst[j, 0:1280] = dec_Wih[j], dst[j, 1280:1792] = dec_Whh[j]
__global__ void k_dec_concat(const float* __restrict__ Wih, const float* __restrict__ Whh,
                             bf16_t* __restrict__ dst) {
  int i = blockIdx.x * 256 + threadIdx.x;
  const int n = H4 * KDEC;
  if (i >= n) return;
  int j = i / KDEC, c = i % KDEC;
  float v = (c < E + 2 * H) ? Wih[(size_t)j * (E + 2 * H) + c]
                            : Whh[(size_t)j * H + (c - (E + 2 * H))];
  dst[i] = (bf16_t)v;
}

__global__ void k_gather_embed(const int* __restrict__ idx, int idx_stride,
                               const float* __restrict__ table,
                               float* __restrict__ out, int rows) {
  int gid = blockIdx.x * 256 + threadIdx.x;
  int n = rows * E;
  if (gid >= n) return;
  int r = gid / E, e = gid % E;
  out[gid] = table[(size_t)idx[(size_t)r * idx_stride] * E + e];
}

// LSTM cell: gates G[B,4H] (i,f,g,o), c update, h out (plus optional mirrors)
__global__ __launch_bounds__(256)
void k_lstm_cell(const float* __restrict__ G, const float* __restrict__ c_in,
                 float* __restrict__ h1, float* __restrict__ c1,
                 float* __restrict__ h2, int ld2h,
                 float* __restrict__ c2, int ld2c) {
  int idx = blockIdx.x * 256 + threadIdx.x;   // [0, B*H)
  int b = idx >> 9, j = idx & (H - 1);
  const float* g = G + (size_t)b * H4;
  float iv = g[j], fv = g[H + j], gv = g[2 * H + j], ov = g[3 * H + j];
  float c = sigf(fv) * c_in[idx] + sigf(iv) * tanhf(gv);
  float h = sigf(ov) * tanhf(c);
  h1[idx] = h;
  c1[idx] = c;
  if (h2) h2[(size_t)b * ld2h + j] = h;
  if (c2) c2[(size_t)b * ld2c + j] = c;
}

// tag self-attention core: qk = K@Q^T/E, softmax over s, *mask + eps,
// w[s] = sum_e a[s,e]; attn[b,s,:] = w[s] * V[b,s,:]
__global__ __launch_bounds__(256)
void k_tag_sa(const float* __restrict__ Kb, const float* __restrict__ Qb,
              const float* __restrict__ Vb, const float* __restrict__ mask,
              float* __restrict__ attn) {
  const int b = blockIdx.x, tid = threadIdx.x;
  __shared__ float qk[Sz][Sz];
  __shared__ float wrow[Sz];
  const float* Kr = Kb + (size_t)b * Sz * H;
  const float* Qr = Qb + (size_t)b * Sz * H;
  for (int p = tid; p < Sz * Sz; p += 256) {
    int s = p / Sz, e = p % Sz;
    const float* kr = Kr + (size_t)s * H;
    const float* qr = Qr + (size_t)e * H;
    float d = 0.f;
    for (int i = 0; i < H; ++i) d += kr[i] * qr[i];
    qk[s][e] = d * (1.0f / (float)E);
  }
  __syncthreads();
  if (tid < Sz) {  // softmax over s (axis 1) per column e
    int e = tid;
    float mx = -1e30f;
    for (int s = 0; s < Sz; ++s) mx = fmaxf(mx, qk[s][e]);
    float sum = 0.f;
    for (int s = 0; s < Sz; ++s) { float ex = __expf(qk[s][e] - mx); qk[s][e] = ex; sum += ex; }
    float inv = 1.0f / sum;
    float mk = mask[(size_t)b * Sz + e];
    for (int s = 0; s < Sz; ++s) qk[s][e] = qk[s][e] * inv * mk + EPSF;
  }
  __syncthreads();
  if (tid < Sz) {
    float w = 0.f;
    for (int e = 0; e < Sz; ++e) w += qk[tid][e];
    wrow[tid] = w;
  }
  __syncthreads();
  for (int p = tid; p < Sz * H; p += 256) {
    int s = p / H, d = p % H;
    size_t o = ((size_t)b * Sz + s) * H + d;
    attn[o] = wrow[s] * Vb[o];
  }
}

// per-row L2 norm over H
__global__ __launch_bounds__(256)
void k_rownorm(const float* __restrict__ A, float* __restrict__ out, int rows) {
  int row = blockIdx.x * 8 + (threadIdx.x >> 5);
  int lane = threadIdx.x & 31;
  if (row >= rows) return;
  const float* r = A + (size_t)row * H;
  float s = 0.f;
  for (int d = lane; d < H; d += 32) s += r[d] * r[d];
  s = wave_reduce_sum(s);
  if (lane == 0) out[row] = sqrtf(s);
}

// CosBah attention over lemma_s: cos part + Bahdanau part -> xcat[B,2H]
__global__ __launch_bounds__(256)
void k_cosbah(const float* __restrict__ hx, const float* __restrict__ kwa, int ldk,
              const float* __restrict__ xs, const float* __restrict__ xswb,
              const float* __restrict__ xnorm, const float* __restrict__ energy,
              const float* __restrict__ mask, float* __restrict__ xcat) {
  const int b = blockIdx.x, tid = threadIdx.x;
  const int lane = tid & 31, wave = tid >> 5;
  __shared__ float shk[H], shkwa[H];
  __shared__ float snum[Lz], sbah[Lz], sacos[Lz], sabah[Lz];
  __shared__ float red[256];
  __shared__ float sms[2];
  shk[tid] = hx[(size_t)b * H + tid];
  shk[tid + 256] = hx[(size_t)b * H + tid + 256];
  shkwa[tid] = kwa[(size_t)b * ldk + tid];
  shkwa[tid + 256] = kwa[(size_t)b * ldk + tid + 256];
  red[tid] = shk[tid] * shk[tid] + shk[tid + 256] * shk[tid + 256];
  __syncthreads();
  for (int s2 = 128; s2 > 0; s2 >>= 1) {
    if (tid < s2) red[tid] += red[tid + s2];
    __syncthreads();
  }
  const float knorm = sqrtf(red[0]);

  for (int l = wave; l < Lz; l += 8) {
    const float* xr = xs + ((size_t)b * Lz + l) * H;
    const float* wr = xswb + ((size_t)b * Lz + l) * H;
    float num = 0.f, bah = 0.f;
    for (int d = lane; d < H; d += 32) {
      num += shk[d] * xr[d];
      bah += tanhf(shkwa[d] + wr[d]) * energy[d];
    }
    num = wave_reduce_sum(num);
    bah = wave_reduce_sum(bah);
    if (lane == 0) { snum[l] = num; sbah[l] = bah; }
  }
  __syncthreads();
  if (tid == 0) {
    float mx = -1e30f;
    for (int l = 0; l < Lz; ++l) mx = fmaxf(mx, sbah[l]);
    float sum = 0.f;
    for (int l = 0; l < Lz; ++l) sum += __expf(sbah[l] - mx);
    sms[0] = mx; sms[1] = sum;
  }
  __syncthreads();
  if (tid < Lz) {
    int l = tid;
    float mk = mask[(size_t)b * Lz + l];
    float den = fmaxf(knorm * xnorm[(size_t)b * Lz + l], 1e-8f);
    float cv = snum[l] / den;
    sacos[l] = sqrtf(cv * cv + 1e-3f) * mk + EPSF;           // softabs
    sabah[l] = __expf(sbah[l] - sms[0]) / sms[1] * mk + EPSF;
  }
  __syncthreads();
  float c0 = 0.f, c1 = 0.f, b0 = 0.f, b1 = 0.f;
  for (int l = 0; l < Lz; ++l) {
    const float* xr = xs + ((size_t)b * Lz + l) * H;
    float ac = sacos[l], ab = sabah[l];
    c0 += ac * xr[tid];  c1 += ac * xr[tid + 256];
    b0 += ab * xr[tid];  b1 += ab * xr[tid + 256];
  }
  float* o = xcat + (size_t)b * (2 * H);
  o[tid] = c0;        o[tid + 256] = c1;
  o[H + tid] = b0;    o[H + tid + 256] = b1;
}

// Bahdanau attention over tags_h -> out[B,H]
__global__ __launch_bounds__(256)
void k_tag_bah(const float* __restrict__ kwa, int ldk, const float* __restrict__ xs,
               const float* __restrict__ xswb, const float* __restrict__ energy,
               const float* __restrict__ mask, float* __restrict__ out) {
  const int b = blockIdx.x, tid = threadIdx.x;
  const int lane = tid & 31, wave = tid >> 5;
  __shared__ float shkwa[H];
  __shared__ float ssc[Sz], sa[Sz];
  __shared__ float sms[2];
  shkwa[tid] = kwa[(size_t)b * ldk + tid];
  shkwa[tid + 256] = kwa[(size_t)b * ldk + tid + 256];
  __syncthreads();
  for (int s = wave; s < Sz; s += 8) {
    const float* wr = xswb + ((size_t)b * Sz + s) * H;
    float acc = 0.f;
    for (int d = lane; d < H; d += 32) acc += tanhf(shkwa[d] + wr[d]) * energy[d];
    acc = wave_reduce_sum(acc);
    if (lane == 0) ssc[s] = acc;
  }
  __syncthreads();
  if (tid == 0) {
    float mx = -1e30f;
    for (int s = 0; s < Sz; ++s) mx = fmaxf(mx, ssc[s]);
    float sum = 0.f;
    for (int s = 0; s < Sz; ++s) sum += __expf(ssc[s] - mx);
    sms[0] = mx; sms[1] = sum;
  }
  __syncthreads();
  if (tid < Sz)
    sa[tid] = __expf(ssc[tid] - sms[0]) / sms[1] * mask[(size_t)b * Sz + tid] + EPSF;
  __syncthreads();
  float a0 = 0.f, a1 = 0.f;
  for (int s = 0; s < Sz; ++s) {
    const float* xr = xs + ((size_t)b * Sz + s) * H;
    a0 += sa[s] * xr[tid];
    a1 += sa[s] * xr[tid + 256];
  }
  out[(size_t)b * H + tid] = a0;
  out[(size_t)b * H + tid + 256] = a1;
}

// xin = [x(256) | seq(512) | tag(512) | h(512)]
__global__ void k_concat_xin(const float* __restrict__ x, const float* __restrict__ sq,
                             const float* __restrict__ tg, const float* __restrict__ h,
                             float* __restrict__ xin) {
  int gid = blockIdx.x * 256 + threadIdx.x;  // B*KDEC exact
  int b = gid / KDEC, c = gid % KDEC;
  float v;
  if (c < E)            v = x[(size_t)b * E + c];
  else if (c < E + H)   v = sq[(size_t)b * H + (c - E)];
  else if (c < E + 2*H) v = tg[(size_t)b * H + (c - E - H)];
  else                  v = h[(size_t)b * H + (c - E - 2 * H)];
  xin[gid] = v;
}

// skinny N=4 projection (ops head)
__global__ void k_skinny4(const float* __restrict__ A, int lda,
                          const float* __restrict__ W,   // [4,K] f32
                          float* __restrict__ out, int out_stride,
                          int rows, int K) {
  int gid = blockIdx.x * 256 + threadIdx.x;
  if (gid >= rows * OPS) return;
  int r = gid >> 2, o = gid & 3;
  const float* a = A + (size_t)r * lda;
  const float* w = W + (size_t)o * K;
  float s = 0.f;
  for (int k = 0; k < K; ++k) s += a[k] * w[k];
  out[(size_t)r * out_stride + o] = s;
}

// greedy feedback: argmax over V (first max wins) + embedding gather
__global__ __launch_bounds__(256)
void k_argmax_embed(const float* __restrict__ logits_base, size_t row_stride,
                    const float* __restrict__ emb, float* __restrict__ xcur) {
  int b = blockIdx.x, tid = threadIdx.x;
  const float* lg = logits_base + (size_t)b * row_stride;
  float bv = -1e30f; int bi = 0;
  for (int i = tid * 4; i < tid * 4 + 4; ++i) {
    float v = lg[i];
    if (v > bv) { bv = v; bi = i; }
  }
  __shared__ float sv[256];
  __shared__ int   si[256];
  sv[tid] = bv; si[tid] = bi;
  __syncthreads();
  for (int s = 128; s > 0; s >>= 1) {
    if (tid < s) {
      if (sv[tid + s] > sv[tid] || (sv[tid + s] == sv[tid] && si[tid + s] < si[tid])) {
        sv[tid] = sv[tid + s]; si[tid] = si[tid + s];
      }
    }
    __syncthreads();
  }
  int y = si[0];
  xcur[(size_t)b * E + tid] = emb[(size_t)y * E + tid];  // E == blockDim
}

// ---------------------------------------------------------------------------
// host orchestration
// ---------------------------------------------------------------------------
static void gemm(hipStream_t st, int act,
                 const float* A, int lda, const bf16_t* W,
                 const float* bias, const float* addend, int ldadd,
                 float* C, int ldc, int M, int N, int K) {
  dim3 grid(N / 128, M / 64);
  if (act == 0)      k_gemm<0><<<grid, 256, 0, st>>>(A, lda, W, bias, addend, ldadd, C, ldc, M, N, K);
  else if (act == 1) k_gemm<1><<<grid, 256, 0, st>>>(A, lda, W, bias, addend, ldadd, C, ldc, M, N, K);
  else               k_gemm<2><<<grid, 256, 0, st>>>(A, lda, W, bias, addend, ldadd, C, ldc, M, N, K);
}

extern "C" void kernel_launch(void* const* d_in, const int* in_sizes, int n_in,
                              void* d_out, int out_size, void* d_ws, size_t ws_size,
                              hipStream_t stream) {
  (void)in_sizes; (void)n_in; (void)out_size;

  const int*   lemma      = (const int*)  d_in[0];
  const float* lemma_mask = (const float*)d_in[1];
  const int*   tags       = (const int*)  d_in[2];
  const float* tags_mask  = (const float*)d_in[3];
  // d_in[4] = gold (only its shape matters; TDEC is a compile-time constant)
  const float* char_emb   = (const float*)d_in[5];
  const float* tag_emb    = (const float*)d_in[6];
  const float* enc_Wih_f  = (const float*)d_in[7];
  const float* enc_bih_f  = (const float*)d_in[8];
  const float* enc_Whh_f  = (const float*)d_in[9];
  const float* enc_bhh_f  = (const float*)d_in[10];
  const float* enc_Wih_b  = (const float*)d_in[11];
  const float* enc_bih_b  = (const float*)d_in[12];
  const float* enc_Whh_b  = (const float*)d_in[13];
  const float* enc_bhh_b  = (const float*)d_in[14];
  const float* sa_Wk      = (const float*)d_in[15];
  const float* sa_bk      = (const float*)d_in[16];
  const float* sa_Wq      = (const float*)d_in[17];
  const float* sa_bq      = (const float*)d_in[18];
  const float* sa_Wv      = (const float*)d_in[19];
  const float* sa_bv      = (const float*)d_in[20];
  const float* sa_Wl      = (const float*)d_in[21];
  const float* sa_bl      = (const float*)d_in[22];
  const float* ops_W      = (const float*)d_in[23];
  const float* cls_W1     = (const float*)d_in[24];
  const float* cls_b1     = (const float*)d_in[25];
  const float* cls_W2     = (const float*)d_in[26];
  const float* sc_W       = (const float*)d_in[27];
  const float* sc_b       = (const float*)d_in[28];
  const float* dec_Wih    = (const float*)d_in[29];
  const float* dec_bih    = (const float*)d_in[30];
  const float* dec_Whh    = (const float*)d_in[31];
  const float* dec_bhh    = (const float*)d_in[32];
  const float* ca_energy  = (const float*)d_in[33];
  const float* ca_Wa      = (const float*)d_in[34];
  const float* ca_ba      = (const float*)d_in[35];
  const float* ca_Wb      = (const float*)d_in[36];
  const float* ca_bb      = (const float*)d_in[37];
  const float* ca_Wc      = (const float*)d_in[38];
  const float* ca_bc      = (const float*)d_in[39];
  const float* ta_energy  = (const float*)d_in[40];
  const float* ta_Wa      = (const float*)d_in[41];
  const float* ta_ba      = (const float*)d_in[42];
  const float* ta_Wb      = (const float*)d_in[43];
  const float* ta_bb      = (const float*)d_in[44];

  float* outF = (float*)d_out;
  const size_t DECN  = (size_t)Bz * Tz * V;
  const size_t OPSYN = (size_t)Bz * Tz * OPS;
  float* out_dec  = outF;
  float* out_opsy = outF + DECN;
  float* out_opsx = outF + DECN + OPSYN;

  // ---- workspace bump allocator ----
  size_t off = 0;
  auto alloc = [&](size_t nbytes) -> void* {
    void* p = (char*)d_ws + off;
    off += (nbytes + 255) & ~(size_t)255;
    return p;
  };
  // bf16 weights
  bf16_t* wWihF  = (bf16_t*)alloc((size_t)H4 * E * 2);
  bf16_t* wWhhF  = (bf16_t*)alloc((size_t)H4 * H * 2);
  bf16_t* wWihB  = (bf16_t*)alloc((size_t)H4 * E * 2);
  bf16_t* wWhhB  = (bf16_t*)alloc((size_t)H4 * H * 2);
  bf16_t* wSaK   = (bf16_t*)alloc((size_t)H * E * 2);
  bf16_t* wSaQ   = (bf16_t*)alloc((size_t)H * E * 2);
  bf16_t* wSaV   = (bf16_t*)alloc((size_t)H * E * 2);
  bf16_t* wSaL   = (bf16_t*)alloc((size_t)H * H * 2);
  bf16_t* wClsW1 = (bf16_t*)alloc((size_t)(2*H) * (2*H) * 2);
  bf16_t* wClsW2 = (bf16_t*)alloc((size_t)V * (2*H) * 2);
  bf16_t* wScW   = (bf16_t*)alloc((size_t)H * (2*H) * 2);
  bf16_t* wAttnA = (bf16_t*)alloc((size_t)(2*H) * H * 2);  // [ca_Wa ; ta_Wa]
  bf16_t* wCaWb  = (bf16_t*)alloc((size_t)H * H * 2);
  bf16_t* wCaWc  = (bf16_t*)alloc((size_t)H * (2*H) * 2);
  bf16_t* wTaWb  = (bf16_t*)alloc((size_t)H * H * 2);
  bf16_t* wDec   = (bf16_t*)alloc((size_t)H4 * KDEC * 2);
  // f32 persistent
  float* biasF   = (float*)alloc((size_t)H4 * 4);
  float* biasB   = (float*)alloc((size_t)H4 * 4);
  float* biasD   = (float*)alloc((size_t)H4 * 4);
  float* biasAtt = (float*)alloc((size_t)2 * H * 4);       // [ca_ba ; ta_ba]
  float* lemma_s = (float*)alloc((size_t)Bz * Lz * H * 4);
  float* xswb    = (float*)alloc((size_t)Bz * Lz * H * 4);
  float* tags_h  = (float*)alloc((size_t)Bz * Sz * H * 4);
  float* tawb    = (float*)alloc((size_t)Bz * Sz * H * 4);
  float* xnorm   = (float*)alloc((size_t)Bz * Lz * 4);
  float* G       = (float*)alloc((size_t)Bz * H4 * 4);
  float* h_cur   = (float*)alloc((size_t)Bz * H * 4);
  float* c_cur   = (float*)alloc((size_t)Bz * H * 4);
  float* kwa2    = (float*)alloc((size_t)Bz * 2 * H * 4);  // fused [kWa_ca | kWa_ta]
  float* xcat    = (float*)alloc((size_t)Bz * 2 * H * 4);
  float* seqA    = (float*)alloc((size_t)Bz * H * 4);
  float* tagA    = (float*)alloc((size_t)Bz * H * 4);
  float* xin     = (float*)alloc((size_t)Bz * KDEC * 4);
  float* hcx     = (float*)alloc((size_t)Bz * 2 * H * 4);
  float* h1buf   = (float*)alloc((size_t)Bz * 2 * H * 4);
  float* x_cur   = (float*)alloc((size_t)Bz * E * 4);
  // transient region reused by tag phase then lemma phase
  const size_t TRANS_F = (size_t)Bz * Lz * E + (size_t)Bz * Lz * 2 * H; // >= tag needs
  float* trans = (float*)alloc(TRANS_F * 4);
  if (off > ws_size) return;  // insufficient scratch; bail deterministically

  // carve transient
  float* tags_e = trans;                                   // [B*S, E]
  float* Kb     = trans + (size_t)Bz * Sz * E;             // [B*S, H]
  float* Qb     = Kb + (size_t)Bz * Sz * H;
  float* Vb     = Qb + (size_t)Bz * Sz * H;
  float* attnb  = Vb + (size_t)Bz * Sz * H;
  float* enc_in  = trans;                                  // [B*L, E]
  float* lemma_h = trans + (size_t)Bz * Lz * E;            // [B*L, 2H]

  auto cvt = [&](const float* s, bf16_t* d, int n) {
    k_cvt_bf16<<<(n + 255) / 256, 256, 0, stream>>>(s, d, n);
  };

  // === Phase 0: weight conversion / fusion ===
  cvt(enc_Wih_f, wWihF, H4 * E);
  cvt(enc_Whh_f, wWhhF, H4 * H);
  cvt(enc_Wih_b, wWihB, H4 * E);
  cvt(enc_Whh_b, wWhhB, H4 * H);
  cvt(sa_Wk, wSaK, H * E);
  cvt(sa_Wq, wSaQ, H * E);
  cvt(sa_Wv, wSaV, H * E);
  cvt(sa_Wl, wSaL, H * H);
  cvt(cls_W1, wClsW1, 2*H * 2*H);
  cvt(cls_W2, wClsW2, V * 2*H);
  cvt(sc_W, wScW, H * 2*H);
  cvt(ca_Wa, wAttnA, H * H);                 // rows 0..511  of fused attn-A weight
  cvt(ta_Wa, wAttnA + (size_t)H * H, H * H); // rows 512..1023
  cvt(ca_Wb, wCaWb, H * H);
  cvt(ca_Wc, wCaWc, H * 2*H);
  cvt(ta_Wb, wTaWb, H * H);
  k_dec_concat<<<(H4 * KDEC + 255) / 256, 256, 0, stream>>>(dec_Wih, dec_Whh, wDec);
  k_add2<<<(H4 + 255) / 256, 256, 0, stream>>>(enc_bih_f, enc_bhh_f, biasF, H4);
  k_add2<<<(H4 + 255) / 256, 256, 0, stream>>>(enc_bih_b, enc_bhh_b, biasB, H4);
  k_add2<<<(H4 + 255) / 256, 256, 0, stream>>>(dec_bih, dec_bhh, biasD, H4);
  hipMemcpyAsync(biasAtt,     ca_ba, (size_t)H * 4, hipMemcpyDeviceToDevice, stream);
  hipMemcpyAsync(biasAtt + H, ta_ba, (size_t)H * 4, hipMemcpyDeviceToDevice, stream);

  // === Phase 1: tag encoder (uses transient region) ===
  k_gather_embed<<<(Bz * Sz * E + 255) / 256, 256, 0, stream>>>(tags, 1, tag_emb, tags_e, Bz * Sz);
  gemm(stream, 0, tags_e, E, wSaK, sa_bk, nullptr, 0, Kb, H, Bz * Sz, H, E);
  gemm(stream, 0, tags_e, E, wSaQ, sa_bq, nullptr, 0, Qb, H, Bz * Sz, H, E);
  gemm(stream, 0, tags_e, E, wSaV, sa_bv, nullptr, 0, Vb, H, Bz * Sz, H, E);
  k_tag_sa<<<Bz, 256, 0, stream>>>(Kb, Qb, Vb, tags_mask, attnb);
  gemm(stream, 2, attnb, H, wSaL, sa_bl, nullptr, 0, tags_h, H, Bz * Sz, H, H);  // leaky relu
  gemm(stream, 0, tags_h, H, wTaWb, ta_bb, nullptr, 0, tawb, H, Bz * Sz, H, H);  // hoisted xs@Wb+bb

  // === Phase 2: biLSTM lemma encoder (reuses transient region) ===
  k_gather_embed<<<(Bz * Lz * E + 255) / 256, 256, 0, stream>>>(lemma, 1, char_emb, enc_in, Bz * Lz);
  hipMemsetAsync(h_cur, 0, (size_t)Bz * H * 4, stream);
  hipMemsetAsync(c_cur, 0, (size_t)Bz * H * 4, stream);
  for (int t = 0; t < Lz; ++t) {  // forward
    gemm(stream, 0, enc_in + (size_t)t * E, Lz * E, wWihF, biasF, nullptr, 0, G, H4, Bz, H4, E);
    gemm(stream, 0, h_cur, H, wWhhF, nullptr, G, H4, G, H4, Bz, H4, H);
    k_lstm_cell<<<Bz * H / 256, 256, 0, stream>>>(G, c_cur, h_cur, c_cur,
        lemma_h + (size_t)t * 2 * H, Lz * 2 * H, nullptr, 0);
  }
  hipMemsetAsync(h_cur, 0, (size_t)Bz * H * 4, stream);
  hipMemsetAsync(c_cur, 0, (size_t)Bz * H * 4, stream);
  for (int tt = 0; tt < Lz; ++tt) {  // backward
    int t = Lz - 1 - tt;
    gemm(stream, 0, enc_in + (size_t)t * E, Lz * E, wWihB, biasB, nullptr, 0, G, H4, Bz, H4, E);
    gemm(stream, 0, h_cur, H, wWhhB, nullptr, G, H4, G, H4, Bz, H4, H);
    k_lstm_cell<<<Bz * H / 256, 256, 0, stream>>>(G, c_cur, h_cur, c_cur,
        lemma_h + (size_t)t * 2 * H + H, Lz * 2 * H, nullptr, 0);
  }
  // heads off lemma_h
  k_skinny4<<<(Bz * Lz * OPS + 255) / 256, 256, 0, stream>>>(lemma_h, 2 * H, ops_W,
                                                             out_opsx, OPS, Bz * Lz, 2 * H);
  gemm(stream, 0, lemma_h, 2 * H, wScW, sc_b, nullptr, 0, lemma_s, H, Bz * Lz, H, 2 * H);
  gemm(stream, 0, lemma_s, H, wCaWb, ca_bb, nullptr, 0, xswb, H, Bz * Lz, H, H);  // hoisted
  k_rownorm<<<Bz * Lz / 8, 256, 0, stream>>>(lemma_s, xnorm, Bz * Lz);

  // === Phase 3: greedy decoder, TDEC steps ===
  k_gather_embed<<<(Bz * E + 255) / 256, 256, 0, stream>>>(lemma, Lz, char_emb, x_cur, Bz);
  hipMemsetAsync(h_cur, 0, (size_t)Bz * H * 4, stream);
  hipMemsetAsync(c_cur, 0, (size_t)Bz * H * 4, stream);
  for (int t = 0; t < Tz; ++t) {
    // fused attention key projection: [hx@ca_Wa+ca_ba | hx@ta_Wa+ta_ba]
    gemm(stream, 0, h_cur, H, wAttnA, biasAtt, nullptr, 0, kwa2, 2 * H, Bz, 2 * H, H);
    k_cosbah<<<Bz, 256, 0, stream>>>(h_cur, kwa2, 2 * H, lemma_s, xswb, xnorm,
                                     ca_energy, lemma_mask, xcat);
    gemm(stream, 0, xcat, 2 * H, wCaWc, ca_bc, nullptr, 0, seqA, H, Bz, H, 2 * H);
    k_tag_bah<<<Bz, 256, 0, stream>>>(kwa2 + H, 2 * H, tags_h, tawb,
                                      ta_energy, tags_mask, tagA);
    k_concat_xin<<<Bz * KDEC / 256, 256, 0, stream>>>(x_cur, seqA, tagA, h_cur, xin);
    gemm(stream, 0, xin, KDEC, wDec, biasD, nullptr, 0, G, H4, Bz, H4, KDEC);
    k_lstm_cell<<<Bz * H / 256, 256, 0, stream>>>(G, c_cur, h_cur, c_cur,
                                                  hcx, 2 * H, hcx + H, 2 * H);
    gemm(stream, 1, hcx, 2 * H, wClsW1, cls_b1, nullptr, 0, h1buf, 2 * H, Bz, 2 * H, 2 * H);
    gemm(stream, 0, h1buf, 2 * H, wClsW2, nullptr, nullptr, 0,
         out_dec + (size_t)t * V, Tz * V, Bz, V, 2 * H);
    k_skinny4<<<(Bz * OPS + 255) / 256, 256, 0, stream>>>(hcx, 2 * H, ops_W,
        out_opsy + (size_t)t * OPS, Tz * OPS, Bz, 2 * H);
    k_argmax_embed<<<Bz, 256, 0, stream>>>(out_dec + (size_t)t * V, (size_t)Tz * V,
                                           char_emb, x_cur);
  }
}
